// MultiHeadLatentAttention_7267084665358
// MI455X (gfx1250) — matrix operand order repaired and checked by reading, not compiled, not run on hardware
//
#include <hip/hip_runtime.h>
#include <stdint.h>

// ---------------------------------------------------------------------------
// MLA forward for MI455X (gfx1250, wave32, WMMA bf16 16x16x32, f32 accum).
//
// Algebra: k,v only appear as (k+v)@W_down -> fold to W_dk = (W_k+W_v)@W_down.
//   latent = nan_to_num((x @ W_dk + memory) * 0.5)        [output 2]
//   q      = x @ W_q
//   per (batch,head): k_h = latent_b @ W_up_k[:,h], v_h likewise,
//   attn = softmax(nan_to_num(q_h k_h^T / 32)), ctx = attn v_h
//   out  = ctx @ W_o                                      [output 1]
//
// GEMMs: block = 8 waves; wave tile 16x64 (4 accumulators). The 64x32 B tile
// is shared by all 8 waves: double-buffered in LDS via
// GLOBAL_LOAD_ASYNC_TO_LDS_B128 (ASYNCcnt), overlapping the next tile's DMA
// with the current tile's WMMAs.  k-loop kept rolled (#pragma unroll 1) so
// accumulator phis coalesce into fixed registers (no copy/nop churn).
// ---------------------------------------------------------------------------

typedef __attribute__((ext_vector_type(16))) __bf16        v16bf;
typedef __attribute__((ext_vector_type(8)))  float         v8f;
typedef __attribute__((ext_vector_type(8)))  unsigned int  v8u;

#define D_MODEL   1024
#define NUM_HEADS 16
#define D_LATENT  256
#define HEAD_DIM  64
#define BATCH     2048
#define SEQ       32
#define NROWS     ((size_t)BATCH * SEQ)   // 65536

__device__ __forceinline__ unsigned short f2bf(float f) {
  unsigned int u = __float_as_uint(f);
  if ((u & 0x7fffffffu) > 0x7f800000u) return (unsigned short)0x7fc0u;  // NaN
  u += 0x7fffu + ((u >> 16) & 1u);                                      // RNE
  return (unsigned short)(u >> 16);
}

__device__ __forceinline__ float nan_to_num_latent(float y) {
  unsigned int u = __float_as_uint(y), a = u & 0x7fffffffu;
  if (a > 0x7f800000u) return 0.0f;
  if (a == 0x7f800000u) return (u >> 31) ? -3.402823466e38f : 3.402823466e38f;
  return y;
}

__device__ __forceinline__ float nan_to_num_score(float y) {
  unsigned int u = __float_as_uint(y), a = u & 0x7fffffffu;
  if (a > 0x7f800000u) return 0.0f;
  if (a == 0x7f800000u) return (u >> 31) ? -10.0f : 10.0f;
  return y;
}

__device__ __forceinline__ v8f wmma_bf16(v16bf a, v16bf b, v8f c) {
  return __builtin_amdgcn_wmma_f32_16x16x32_bf16(false, a, false, b,
                                                 (short)0, c, false, false);
}

// A fragment (16x32 bf16, MxK) from row-major [*, lda] array.
// lane L: m=L&15, h=L>>4; elems 0..7 -> K=k0+8h+i ; elems 8..15 -> K=k0+16+8h+i
__device__ __forceinline__ v16bf load_a_frag(const unsigned short* A, int lda,
                                             int row0, int k0) {
  const int lane = threadIdx.x & 31;
  const int m = lane & 15, h = lane >> 4;
  const unsigned short* p = A + (size_t)(row0 + m) * lda + (k0 + 8 * h);
  uint4 c0 = *(const uint4*)(p);
  uint4 c1 = *(const uint4*)(p + 16);
  v8u u;
  u[0] = c0.x; u[1] = c0.y; u[2] = c0.z; u[3] = c0.w;
  u[4] = c1.x; u[5] = c1.y; u[6] = c1.z; u[7] = c1.w;
  return __builtin_bit_cast(v16bf, u);
}

// B fragment (32x16 bf16, KxN) from a [N][ldk] row-major array (i.e. B^T).
// lane L: n=L&15, h=L>>4; elems 0..7 -> K=k0+8h+i ; elems 8..15 -> K=k0+16+8h+i
__device__ __forceinline__ v16bf load_b_frag(const unsigned short* Wt, int ldk,
                                             int n0, int k0) {
  const int lane = threadIdx.x & 31;
  const int n = lane & 15, h = lane >> 4;
  const unsigned short* p = Wt + (size_t)(n0 + n) * ldk + (k0 + 8 * h);
  uint4 c0 = *(const uint4*)(p);
  uint4 c1 = *(const uint4*)(p + 16);
  v8u u;
  u[0] = c0.x; u[1] = c0.y; u[2] = c0.z; u[3] = c0.w;
  u[4] = c1.x; u[5] = c1.y; u[6] = c1.z; u[7] = c1.w;
  return __builtin_bit_cast(v16bf, u);
}

// ---------------------------------------------------------------------------
// Double-buffered GEMM core.  Block = 256 thr / 8 waves, block tile 128x64.
// Wave w owns rows [row0, row0+16).  B tile (64 cols x 32 ks, [n][k] layout,
// 4 KB) is fetched once per block per k-step with async LDS DMA and consumed
// by all waves via ds_load_b128 fragments.
// ---------------------------------------------------------------------------
#define BK 32
#define BN 64

__device__ __forceinline__ void gemm_tile_16x64(
    const unsigned short* __restrict__ A, int K,
    const unsigned short* __restrict__ Wt,   // [N][K] bf16
    int row0, int col0, unsigned short* bsm /* 2 * BN*BK */, v8f acc[4]) {
  const int t = threadIdx.x;
  const int bn = t >> 2;            // 0..63 : tile-local column (row of Wt)
  const int bj = (t & 3) * 8;       // 0..24 : k offset (8 bf16 = one b128)

  // Issue this thread's b128 of the B tile for k-step k0 into buffer `buf`
  // via async global->LDS DMA (tracked by ASYNCcnt).
  auto issue_b = [&](int buf, int k0) {
    const unsigned short* g = Wt + (size_t)(col0 + bn) * K + (k0 + bj);
    unsigned lds = (unsigned)(uintptr_t)(bsm + buf * (BN * BK) + bn * BK + bj);
    asm volatile("global_load_async_to_lds_b128 %0, %1, off"
                 :: "v"(lds), "v"(g) : "memory");
  };

  issue_b(0, 0);
  const int nsteps = K / BK;
#pragma unroll 1
  for (int s = 0; s < nsteps; ++s) {
    const int cur = s & 1;
    if (s + 1 < nsteps) {
      issue_b(cur ^ 1, (s + 1) * BK);
      asm volatile("s_wait_asynccnt 1" ::: "memory");  // current tile landed
    } else {
      asm volatile("s_wait_asynccnt 0" ::: "memory");
    }
    __syncthreads();
    const unsigned short* bbuf = bsm + cur * (BN * BK);
    v16bf a = load_a_frag(A, K, row0, s * BK);
#pragma unroll
    for (int j = 0; j < 4; ++j) {
      v16bf b = load_b_frag(bbuf, BK, j * 16, 0);
      acc[j] = wmma_bf16(a, b, acc[j]);
    }
    __syncthreads();   // all waves done reading before buffer reuse
  }
}

// ---------------------------------------------------------------------------
// Prep kernels (run once per launch; tiny vs. the GEMMs)
// ---------------------------------------------------------------------------

__global__ void mla_conv_bf16(const float* __restrict__ x,
                              unsigned short* __restrict__ xb, size_t n) {
  size_t i = (size_t)blockIdx.x * blockDim.x + threadIdx.x;
  if (i < n) xb[i] = f2bf(x[i]);
}

// W [R][C] f32 -> Wt [C][R] bf16
__global__ void mla_transpose_bf16(const float* __restrict__ W,
                                   unsigned short* __restrict__ Wt,
                                   int R, int C) {
  int idx = blockIdx.x * blockDim.x + threadIdx.x;
  if (idx < R * C) {
    int r = idx / C, c = idx % C;
    Wt[(size_t)c * R + r] = f2bf(W[(size_t)r * C + c]);
  }
}

// W_dk_t[l][d] = sum_m (W_k[d][m]+W_v[d][m]) * W_down[m][l]   (bf16, [256][1024])
__global__ void mla_prep_wdk(const float* __restrict__ Wk,
                             const float* __restrict__ Wv,
                             const float* __restrict__ Wdown,
                             unsigned short* __restrict__ Wdk_t) {
  int idx = blockIdx.x * blockDim.x + threadIdx.x;   // D_MODEL * D_LATENT threads
  int lcol = idx & (D_LATENT - 1);
  int d    = idx >> 8;
  float s = 0.0f;
  for (int m = 0; m < D_MODEL; ++m)
    s = fmaf(Wk[(size_t)d * D_MODEL + m] + Wv[(size_t)d * D_MODEL + m],
             Wdown[(size_t)m * D_LATENT + lcol], s);
  Wdk_t[(size_t)lcol * D_MODEL + d] = f2bf(s);
}

// ---------------------------------------------------------------------------
// GEMM kernels (block tile 128x64, LDS double-buffered B)
// ---------------------------------------------------------------------------

// q_bf = x_bf @ W_q (via Wq_t [N][K]), bf16 out.  M=65536, N=K=1024.
__global__ __launch_bounds__(256, 1) void mla_gemm_q(
    const unsigned short* __restrict__ A, const unsigned short* __restrict__ Wt,
    unsigned short* __restrict__ Cout) {
  __shared__ unsigned short bsm[2 * BN * BK];
  const int wave = threadIdx.x >> 5;
  const int row0 = blockIdx.y * 128 + wave * 16;
  const int col0 = blockIdx.x * BN;
  v8f acc[4] = {};
  gemm_tile_16x64(A, D_MODEL, Wt, row0, col0, bsm, acc);
  const int lane = threadIdx.x & 31, n = lane & 15, h = lane >> 4;
#pragma unroll
  for (int j = 0; j < 4; ++j)
#pragma unroll
    for (int r = 0; r < 8; ++r) {
      size_t row = (size_t)row0 + r + 8 * h;
      Cout[row * D_MODEL + col0 + 16 * j + n] = f2bf(acc[j][r]);
    }
}

// latent = nan_to_num((x_bf @ W_dk + memory) * 0.5); f32 -> d_out, bf16 -> ws.
// M=65536, N=256, K=1024.
__global__ __launch_bounds__(256, 1) void mla_gemm_latent(
    const unsigned short* __restrict__ A, const unsigned short* __restrict__ Wt,
    const float* __restrict__ memory, float* __restrict__ lat_f32,
    unsigned short* __restrict__ lat_bf) {
  __shared__ unsigned short bsm[2 * BN * BK];
  const int wave = threadIdx.x >> 5;
  const int row0 = blockIdx.y * 128 + wave * 16;
  const int col0 = blockIdx.x * BN;
  v8f acc[4] = {};
  gemm_tile_16x64(A, D_MODEL, Wt, row0, col0, bsm, acc);
  const int lane = threadIdx.x & 31, n = lane & 15, h = lane >> 4;
#pragma unroll
  for (int j = 0; j < 4; ++j)
#pragma unroll
    for (int r = 0; r < 8; ++r) {
      size_t row = (size_t)row0 + r + 8 * h;
      size_t idx = row * D_LATENT + col0 + 16 * j + n;
      float y = nan_to_num_latent((acc[j][r] + memory[idx]) * 0.5f);
      lat_f32[idx] = y;
      lat_bf[idx]  = f2bf(y);
    }
}

// out = ctx_bf @ W_o (via Wo_t), f32 out.  M=65536, N=K=1024.
__global__ __launch_bounds__(256, 1) void mla_gemm_out(
    const unsigned short* __restrict__ A, const unsigned short* __restrict__ Wt,
    float* __restrict__ Cout) {
  __shared__ unsigned short bsm[2 * BN * BK];
  const int wave = threadIdx.x >> 5;
  const int row0 = blockIdx.y * 128 + wave * 16;
  const int col0 = blockIdx.x * BN;
  v8f acc[4] = {};
  gemm_tile_16x64(A, D_MODEL, Wt, row0, col0, bsm, acc);
  const int lane = threadIdx.x & 31, n = lane & 15, h = lane >> 4;
#pragma unroll
  for (int j = 0; j < 4; ++j)
#pragma unroll
    for (int r = 0; r < 8; ++r) {
      size_t row = (size_t)row0 + r + 8 * h;
      Cout[row * D_MODEL + col0 + 16 * j + n] = acc[j][r];
    }
}

// ---------------------------------------------------------------------------
// Fused attention: 1 block / batch, 1 wave / head.  Per-wave LDS slot:
//   kh [32][64] bf16 (4 KB, reused as attn [32][32]) | vt [64][32] bf16 (4 KB)
//   sc [32][32] f32  (4 KB)            -> 12 KB/wave * 16 waves = 192 KB/WGP
// ---------------------------------------------------------------------------
#define ATTN_SLOT 12288

__global__ __launch_bounds__(512) void mla_attention(
    const unsigned short* __restrict__ q_bf,
    const unsigned short* __restrict__ lat_bf,
    const unsigned short* __restrict__ Wupk_t,   // [1024][256]
    const unsigned short* __restrict__ Wupv_t,   // [1024][256]
    unsigned short* __restrict__ ctx_bf) {
  extern __shared__ char attn_smem[];
  const int wave = threadIdx.x >> 5;          // head index
  const int lane = threadIdx.x & 31;
  const int n = lane & 15, h2 = lane >> 4;
  const int row0 = blockIdx.x * SEQ;          // first row of this batch

  unsigned short* kh = (unsigned short*)(attn_smem + wave * ATTN_SLOT);
  unsigned short* vt = (unsigned short*)(attn_smem + wave * ATTN_SLOT + 4096);
  float*          sc = (float*)        (attn_smem + wave * ATTN_SLOT + 8192);
  unsigned short* at = kh;                    // attn reuses kh region

  // --- stage 1: k_h = latent_b @ W_up_k[:,h]; v_h likewise (32x64 each) ---
  for (int nt = 0; nt < 4; ++nt) {
    v8f ak[2] = {}, av[2] = {};
    const int ncol = wave * HEAD_DIM + nt * 16;
    for (int k0 = 0; k0 < D_LATENT; k0 += 32) {
      v16bf a0 = load_a_frag(lat_bf, D_LATENT, row0, k0);
      v16bf a1 = load_a_frag(lat_bf, D_LATENT, row0 + 16, k0);
      v16bf bk = load_b_frag(Wupk_t, D_LATENT, ncol, k0);
      v16bf bv = load_b_frag(Wupv_t, D_LATENT, ncol, k0);
      ak[0] = wmma_bf16(a0, bk, ak[0]);
      ak[1] = wmma_bf16(a1, bk, ak[1]);
      av[0] = wmma_bf16(a0, bv, av[0]);
      av[1] = wmma_bf16(a1, bv, av[1]);
    }
#pragma unroll
    for (int mt = 0; mt < 2; ++mt)
#pragma unroll
      for (int r = 0; r < 8; ++r) {
        int s = mt * 16 + r + 8 * h2;
        kh[s * HEAD_DIM + nt * 16 + n] = f2bf(ak[mt][r]);   // kh[s][d]
        vt[(nt * 16 + n) * SEQ + s]    = f2bf(av[mt][r]);   // v transposed [d][s]
      }
  }
  __syncthreads();

  // --- stage 2: scores = q_h @ kh^T  (32x32, K=64) ---
  {
    v8f s_acc[2][2] = {};
    for (int kk = 0; kk < 2; ++kk) {
      int k0 = kk * 32;
      v16bf aq0 = load_a_frag(q_bf, D_MODEL, row0,      wave * HEAD_DIM + k0);
      v16bf aq1 = load_a_frag(q_bf, D_MODEL, row0 + 16, wave * HEAD_DIM + k0);
      v16bf bk0 = load_b_frag(kh, HEAD_DIM, 0,  k0);
      v16bf bk1 = load_b_frag(kh, HEAD_DIM, 16, k0);
      s_acc[0][0] = wmma_bf16(aq0, bk0, s_acc[0][0]);
      s_acc[0][1] = wmma_bf16(aq0, bk1, s_acc[0][1]);
      s_acc[1][0] = wmma_bf16(aq1, bk0, s_acc[1][0]);
      s_acc[1][1] = wmma_bf16(aq1, bk1, s_acc[1][1]);
    }
#pragma unroll
    for (int mt = 0; mt < 2; ++mt)
#pragma unroll
      for (int ntt = 0; ntt < 2; ++ntt)
#pragma unroll
        for (int r = 0; r < 8; ++r)
          sc[(mt * 16 + r + 8 * h2) * SEQ + ntt * 16 + n] = s_acc[mt][ntt][r];
  }
  __syncthreads();

  // --- stage 3: softmax (lane = query row), scale 1/sqrt(1024)=1/32 ---
  {
    float vals[SEQ];
    float mx = -3.0e38f;
#pragma unroll
    for (int j = 0; j < SEQ; ++j) {
      float xv = nan_to_num_score(sc[lane * SEQ + j] * 0.03125f);
      vals[j] = xv;
      mx = fmaxf(mx, xv);
    }
    float sum = 0.0f;
#pragma unroll
    for (int j = 0; j < SEQ; ++j) {
      vals[j] = __expf(vals[j] - mx);
      sum += vals[j];
    }
    float inv = 1.0f / sum;
#pragma unroll
    for (int j = 0; j < SEQ; ++j)
      at[lane * SEQ + j] = f2bf(vals[j] * inv);   // attn [q][k] bf16 (kh dead)
  }
  __syncthreads();

  // --- stage 4: ctx = attn @ v_h  (32x64, K=32) ---
  {
    v16bf a0 = load_a_frag(at, SEQ, 0,  0);
    v16bf a1 = load_a_frag(at, SEQ, 16, 0);
#pragma unroll
    for (int nt = 0; nt < 4; ++nt) {
      v16bf bv = load_b_frag(vt, SEQ, nt * 16, 0);
      v8f c0 = {}, c1 = {};
      c0 = wmma_bf16(a0, bv, c0);
      c1 = wmma_bf16(a1, bv, c1);
#pragma unroll
      for (int r = 0; r < 8; ++r) {
        size_t rr0 = (size_t)(row0 + r + 8 * h2);
        size_t rr1 = (size_t)(row0 + 16 + r + 8 * h2);
        size_t col = (size_t)wave * HEAD_DIM + nt * 16 + n;
        ctx_bf[rr0 * D_MODEL + col] = f2bf(c0[r]);
        ctx_bf[rr1 * D_MODEL + col] = f2bf(c1[r]);
      }
    }
  }
}

// ---------------------------------------------------------------------------
// Host launcher
// ---------------------------------------------------------------------------
extern "C" void kernel_launch(void* const* d_in, const int* in_sizes, int n_in,
                              void* d_out, int out_size, void* d_ws,
                              size_t ws_size, hipStream_t stream) {
  const float* x      = (const float*)d_in[0];
  const float* memory = (const float*)d_in[1];
  const float* W_q    = (const float*)d_in[2];
  const float* W_k    = (const float*)d_in[3];
  const float* W_v    = (const float*)d_in[4];
  const float* W_o    = (const float*)d_in[5];
  const float* W_down = (const float*)d_in[6];
  const float* W_up_k = (const float*)d_in[7];
  const float* W_up_v = (const float*)d_in[8];

  float* out_f32 = (float*)d_out;                       // [65536,1024]
  float* lat_f32 = out_f32 + NROWS * D_MODEL;           // [65536,256]

  char* ws = (char*)d_ws;
  size_t off = 0;
  unsigned short* x_bf   = (unsigned short*)(ws + off); off += NROWS * D_MODEL * 2;
  unsigned short* q_bf   = (unsigned short*)(ws + off); off += NROWS * D_MODEL * 2;
  unsigned short* ctx_bf = (unsigned short*)(ws + off); off += NROWS * D_MODEL * 2;
  unsigned short* lat_bf = (unsigned short*)(ws + off); off += NROWS * D_LATENT * 2;
  unsigned short* Wq_t   = (unsigned short*)(ws + off); off += (size_t)D_MODEL * D_MODEL * 2;
  unsigned short* Wo_t   = (unsigned short*)(ws + off); off += (size_t)D_MODEL * D_MODEL * 2;
  unsigned short* Wdk_t  = (unsigned short*)(ws + off); off += (size_t)D_LATENT * D_MODEL * 2;
  unsigned short* Wupk_t = (unsigned short*)(ws + off); off += (size_t)D_MODEL * D_LATENT * 2;
  unsigned short* Wupv_t = (unsigned short*)(ws + off); off += (size_t)D_MODEL * D_LATENT * 2;

  // 1) precision conversion + weight prep (bf16, pre-transposed to [N][K])
  {
    size_t nx = NROWS * D_MODEL;
    mla_conv_bf16<<<(unsigned)((nx + 255) / 256), 256, 0, stream>>>(x, x_bf, nx);
    mla_transpose_bf16<<<(D_MODEL * D_MODEL) / 256, 256, 0, stream>>>(W_q, Wq_t, D_MODEL, D_MODEL);
    mla_transpose_bf16<<<(D_MODEL * D_MODEL) / 256, 256, 0, stream>>>(W_o, Wo_t, D_MODEL, D_MODEL);
    mla_transpose_bf16<<<(D_LATENT * D_MODEL) / 256, 256, 0, stream>>>(W_up_k, Wupk_t, D_LATENT, D_MODEL);
    mla_transpose_bf16<<<(D_LATENT * D_MODEL) / 256, 256, 0, stream>>>(W_up_v, Wupv_t, D_LATENT, D_MODEL);
    mla_prep_wdk<<<(D_MODEL * D_LATENT) / 256, 256, 0, stream>>>(W_k, W_v, W_down, Wdk_t);
  }

  // 2) q = x @ W_q        (M=65536, N=1024, K=1024)
  mla_gemm_q<<<dim3(D_MODEL / BN, (unsigned)(NROWS / 128)), 256, 0, stream>>>(x_bf, Wq_t, q_bf);

  // 3) latent = nan_to_num((x @ W_dk + memory) * 0.5)   (N=256)
  mla_gemm_latent<<<dim3(D_LATENT / BN, (unsigned)(NROWS / 128)), 256, 0, stream>>>(
      x_bf, Wdk_t, memory, lat_f32, lat_bf);

  // 4) fused per-batch attention (k/v reconstruction + softmax + ctx)
  mla_attention<<<BATCH, 512, NUM_HEADS * ATTN_SLOT, stream>>>(
      q_bf, lat_bf, Wupk_t, Wupv_t, ctx_bf);

  // 5) out = ctx @ W_o    (M=65536, N=1024, K=1024)
  mla_gemm_out<<<dim3(D_MODEL / BN, (unsigned)(NROWS / 128)), 256, 0, stream>>>(ctx_bf, Wo_t, out_f32);
}
